// HyperConv_70480413327622
// MI455X (gfx1250) — matrix operand order repaired and checked by reading, not compiled, not run on hardware
//
#include <hip/hip_runtime.h>
#include <hip/hip_bf16.h>

// ---------------------------------------------------------------------------
// HyperConv on MI455X (gfx1250): everything is GEMM (bf16 WMMA, fp32 accum).
//  conv1d(KS=2, VALID) == GEMM on the flat input with lda=K0, K=2*K0.
// All B operands are consumed TRANSPOSED (N x K row-major) so both LDS tiles
// stage with contiguous b128 loads/stores and fragments are ds_load_b128.
// Pipeline:
//  z -> zb(bf16); ww*/bw* -> bf16 transposed (N x K)
//  h1w = relu(zb * ww1 + wb1)             M=4097 K=256   (bf16)
//  wts = h1w * ww2 + wb2                  M=4096 K=1024  (bf16, stored F x T)
//  h1b = relu(zb * bw1 + bb1)             M=513  K=256   (bf16; only 512 used)
//  bm  = h1b * bw2 + bb2                  M=512  K=1024  (fp32)
//  out = xu_top * wts^T + bm              M=512  K=4096  (fp32)
// ---------------------------------------------------------------------------

typedef __attribute__((ext_vector_type(16))) __bf16 v16bf;
typedef __attribute__((ext_vector_type(8)))  float  v8f;
typedef __attribute__((ext_vector_type(8)))  unsigned short us8;

static __device__ __forceinline__ unsigned short f2bf(float f) {
    union { float f; unsigned u; } v; v.f = f;
    unsigned r = v.u + 0x7FFFu + ((v.u >> 16) & 1u);   // round-to-nearest-even
    return (unsigned short)(r >> 16);
}

// ------------------------- elementwise helpers -----------------------------

__global__ void hc_f32_to_bf16(const float* __restrict__ in,
                               unsigned short* __restrict__ out, long n) {
    long i = (long)blockIdx.x * blockDim.x + threadIdx.x;
    long stride = (long)gridDim.x * blockDim.x;
    for (; i < n; i += stride) out[i] = f2bf(in[i]);
}

// in: (K x N) f32 row-major  ->  out: (N x K) bf16 row-major
__global__ void hc_f32_to_bf16_T(const float* __restrict__ in,
                                 unsigned short* __restrict__ out,
                                 int K, int N, long n) {
    long i = (long)blockIdx.x * blockDim.x + threadIdx.x;
    long stride = (long)gridDim.x * blockDim.x;
    for (; i < n; i += stride) {
        int k  = (int)(i % K);
        int nn = (int)(i / K);
        out[i] = f2bf(in[(long)k * N + nn]);
    }
}

// xu_top[b,c,t] = x[b,c,t+1] (0 at t = T-1); same flat layout as x.
__global__ void hc_build_xu(const float* __restrict__ x,
                            unsigned short* __restrict__ out, int T, long n) {
    long i = (long)blockIdx.x * blockDim.x + threadIdx.x;
    long stride = (long)gridDim.x * blockDim.x;
    for (; i < n; i += stride) {
        int t = (int)(i % T);
        out[i] = (t == T - 1) ? (unsigned short)0 : f2bf(x[i + 1]);
    }
}

// ------------------------- generic bf16 WMMA GEMM --------------------------
// C[b, m, n] = act( sum_k A[b, m*lda + k] * Bt[b, n*ldbt + k] + bias )
// Block tile 128x128, K-step 32, double-buffered LDS, one barrier/iter.
// 8 waves in 4(M) x 2(N); wave tile 32x64 -> 8 accum frags, 8 WMMA / K-step.
// N is fixed at 512 via gridDim.y == 4.

#define LSTR 40   // LDS row stride (32 + 8 pad) in ushort

__launch_bounds__(256, 2)
__global__ void hc_gemm_bf16_wmma(
    const unsigned short* __restrict__ A,  long lda,  long strideA,
    const unsigned short* __restrict__ Bt, long ldbt, long strideB,
    int M, int K,
    const float* __restrict__ biasv,                        // bias_mode 0: per-n
    const float* __restrict__ biasmat, long strideBias, int ldbias, // mode 1
    int bias_mode, int relu, int out_bf16,
    unsigned short* __restrict__ out16, float* __restrict__ outf,
    long ldcM, long ldcN, long strideC)
{
    __shared__ unsigned short sA[2][128 * LSTR];   // [m][k]
    __shared__ unsigned short sB[2][128 * LSTR];   // [n][k]

    const int b    = blockIdx.z;
    const int m0   = blockIdx.x * 128;
    const int n0   = blockIdx.y * 128;
    const int tid  = threadIdx.x;
    const int lane = tid & 31;
    const int wave = tid >> 5;
    const int wm   = wave & 3;    // * 32 in M
    const int wn   = wave >> 2;   // * 64 in N

    const unsigned short* Ab  = A  + (long)b * strideA;
    const unsigned short* Btb = Bt + (long)b * strideB;

    v8f acc[2][4];
#pragma unroll
    for (int i = 0; i < 2; ++i)
#pragma unroll
        for (int j = 0; j < 4; ++j)
#pragma unroll
            for (int r = 0; r < 8; ++r) acc[i][j][r] = 0.0f;

    // staging: both tiles are 128 rows x 32 k; 2 threads per row.
    const int row = tid >> 1;            // 0..127
    const int col = (tid & 1) * 16;      // 0 / 16

    const int l15 = lane & 15;
    const int kbA = (lane >> 4) << 3;    // A frag K-half select: 0 / 8
    const int kbB = (lane >> 4) << 3;    // B frag K-half select: 0 / 8

    const int ktiles = K >> 5;

    us8 ra0, ra1, rb0, rb1;
    auto gload = [&](int kt) {
        const int k0 = kt << 5;
        const int gm = m0 + row;
        if (gm < M) {
            const us8* p = (const us8*)(Ab + (long)gm * lda + k0 + col);
            ra0 = p[0]; ra1 = p[1];
        } else {
#pragma unroll
            for (int i = 0; i < 8; ++i) { ra0[i] = 0; ra1[i] = 0; }
        }
        const us8* q = (const us8*)(Btb + (long)(n0 + row) * ldbt + k0 + col);
        rb0 = q[0]; rb1 = q[1];
    };
    auto sstore = [&](int buf) {
        *(us8*)(&sA[buf][row * LSTR + col])     = ra0;
        *(us8*)(&sA[buf][row * LSTR + col + 8]) = ra1;
        *(us8*)(&sB[buf][row * LSTR + col])     = rb0;
        *(us8*)(&sB[buf][row * LSTR + col + 8]) = rb1;
    };

    gload(0);
    sstore(0);
    __syncthreads();

    for (int kt = 0; kt < ktiles; ++kt) {
        const int cur = kt & 1;

        // ---- fragments from current buffer ----
        union Frag { v16bf v; us8 h[2]; };
        Frag fa[2], fb[4];
#pragma unroll
        for (int mf = 0; mf < 2; ++mf) {
            const unsigned short* base = &sA[cur][(wm * 32 + mf * 16 + l15) * LSTR];
            fa[mf].h[0] = *(const us8*)(base + kbA);        // K 0..7   / 8..15
            fa[mf].h[1] = *(const us8*)(base + 16 + kbA);   // K 16..23 / 24..31
        }
#pragma unroll
        for (int nf = 0; nf < 4; ++nf) {
            const unsigned short* base =
                &sB[cur][(wn * 64 + nf * 16 + l15) * LSTR + kbB];
            fb[nf].h[0] = *(const us8*)(base);              // K 0..7   / 8..15
            fb[nf].h[1] = *(const us8*)(base + 16);         // K 16..23 / 24..31
        }

        // ---- start next tile's global loads (overlap with WMMA) ----
        if (kt + 1 < ktiles) gload(kt + 1);
        if (kt + 2 < ktiles) {
            const int k0 = (kt + 2) << 5;
            const int gm = m0 + row;
            if (gm < M)
                __builtin_prefetch(Ab + (long)gm * lda + k0 + col, 0, 3);
            __builtin_prefetch(Btb + (long)(n0 + row) * ldbt + k0 + col, 0, 3);
        }

        // ---- 8 WMMAs (A frags reused over 4 N-frags, B over 2 M-frags) ----
#pragma unroll
        for (int mf = 0; mf < 2; ++mf)
#pragma unroll
            for (int nf = 0; nf < 4; ++nf)
                acc[mf][nf] = __builtin_amdgcn_wmma_f32_16x16x32_bf16(
                    false, fa[mf].v, false, fb[nf].v,
                    (short)0, acc[mf][nf], false, false);

        if (kt + 1 < ktiles) sstore(1 - cur);
        __syncthreads();
    }

    // ---- epilogue: C/D layout: VGPR r -> M = r + 8*(lane>=16), N = lane&15
    const int mrow = ((lane >> 4) << 3);
#pragma unroll
    for (int mf = 0; mf < 2; ++mf) {
#pragma unroll
        for (int nf = 0; nf < 4; ++nf) {
            const int gn = n0 + wn * 64 + nf * 16 + l15;
#pragma unroll
            for (int r = 0; r < 8; ++r) {
                const int gm = m0 + wm * 32 + mf * 16 + r + mrow;
                if (gm < M) {
                    float v = acc[mf][nf][r];
                    if (bias_mode == 0)      v += biasv[gn];
                    else if (bias_mode == 1) v += biasmat[(long)b * strideBias +
                                                          (long)gm * ldbias + gn];
                    if (relu) v = v > 0.f ? v : 0.f;
                    const long off = (long)b * strideC + (long)gm * ldcM +
                                     (long)gn * ldcN;
                    if (out_bf16) out16[off] = f2bf(v);
                    else          outf[off]  = v;
                }
            }
        }
    }
}

// --------------------------------- launcher --------------------------------

extern "C" void kernel_launch(void* const* d_in, const int* in_sizes, int n_in,
                              void* d_out, int out_size, void* d_ws, size_t ws_size,
                              hipStream_t stream) {
    (void)in_sizes; (void)n_in; (void)out_size; (void)ws_size;
    const float* x   = (const float*)d_in[0];
    const float* z   = (const float*)d_in[1];
    const float* ww1 = (const float*)d_in[2];
    const float* wb1 = (const float*)d_in[3];
    const float* ww2 = (const float*)d_in[4];
    const float* wb2 = (const float*)d_in[5];
    const float* bw1 = (const float*)d_in[6];
    const float* bb1 = (const float*)d_in[7];
    const float* bw2 = (const float*)d_in[8];
    const float* bb2 = (const float*)d_in[9];
    float* out = (float*)d_out;

    const long Bc = 8, C = 512, T = 4096, F = 512, Z = 128, TZ = 4098;

    char* wsp = (char*)d_ws;
    auto alloc = [&](long bytes) -> char* {
        char* p = wsp; wsp += (bytes + 255) & ~255L; return p;
    };
    unsigned short* zb   = (unsigned short*)alloc(Bc * TZ * Z * 2);     // 8.4 MB
    unsigned short* ww1t = (unsigned short*)alloc(2 * Z * F * 2);       // F x 256
    unsigned short* ww2t = (unsigned short*)alloc(2 * F * F * 2);       // F x 1024
    unsigned short* bw1t = (unsigned short*)alloc(2 * Z * F * 2);
    unsigned short* bw2t = (unsigned short*)alloc(2 * F * F * 2);
    unsigned short* h1w  = (unsigned short*)alloc(Bc * 4097 * F * 2);   // 33.6 MB
    unsigned short* wts  = (unsigned short*)alloc(Bc * F * T * 2);      // F x T !
    unsigned short* h1b  = (unsigned short*)alloc(Bc * 513 * F * 2);    //  4.2 MB
    float*          bm   = (float*)alloc(Bc * C * F * 4);               //  8.4 MB
    unsigned short* xu   = (unsigned short*)alloc(Bc * C * T * 2);      // 33.6 MB

    // conversions (weights transposed to N x K)
    hc_f32_to_bf16<<<1024, 256, 0, stream>>>(z, zb, Bc * TZ * Z);
    hc_f32_to_bf16_T<<<64,  256, 0, stream>>>(ww1, ww1t, 2 * (int)Z, (int)F, 2 * Z * F);
    hc_f32_to_bf16_T<<<256, 256, 0, stream>>>(ww2, ww2t, 2 * (int)F, (int)F, 2 * F * F);
    hc_f32_to_bf16_T<<<64,  256, 0, stream>>>(bw1, bw1t, 2 * (int)Z, (int)F, 2 * Z * F);
    hc_f32_to_bf16_T<<<256, 256, 0, stream>>>(bw2, bw2t, 2 * (int)F, (int)F, 2 * F * F);
    hc_build_xu<<<2048, 256, 0, stream>>>(x, xu, (int)T, Bc * C * T);

    dim3 blk(256);
    // weights adaptor conv1: h1w = relu(zb * ww1 + wb1), M=4097, K=256
    hc_gemm_bf16_wmma<<<dim3(33, 4, 8), blk, 0, stream>>>(
        zb, Z, TZ * Z,  ww1t, 2 * Z, 0,  4097, 256,
        wb1, nullptr, 0, 0,  /*bias_mode*/0, /*relu*/1, /*out_bf16*/1,
        h1w, nullptr, /*ldcM*/F, /*ldcN*/1, 4097 * F);
    // weights adaptor conv2: wts^T (F x T) = (h1w * ww2 + wb2)^T, M=4096, K=1024
    hc_gemm_bf16_wmma<<<dim3(32, 4, 8), blk, 0, stream>>>(
        h1w, F, 4097 * F,  ww2t, 2 * F, 0,  4096, 1024,
        wb2, nullptr, 0, 0,  0, 0, 1,
        wts, nullptr, /*ldcM*/1, /*ldcN*/T, F * T);          // transposed store
    // bias adaptor conv1 (only 513 rows needed): h1b, M=513, K=256
    hc_gemm_bf16_wmma<<<dim3(5, 4, 8), blk, 0, stream>>>(
        zb, Z, TZ * Z,  bw1t, 2 * Z, 0,  513, 256,
        bb1, nullptr, 0, 0,  0, 1, 1,
        h1b, nullptr, F, 1, 513 * F);
    // bias adaptor conv2 (fp32 out): bm, M=512, K=1024
    hc_gemm_bf16_wmma<<<dim3(4, 4, 8), blk, 0, stream>>>(
        h1b, F, 513 * F,  bw2t, 2 * F, 0,  512, 1024,
        bb2, nullptr, 0, 0,  0, 0, 0,
        nullptr, bm, F, 1, C * F);
    // main GEMM: out = xu_top * wts^T + bm, M=512, K=4096 (fp32 out)
    hc_gemm_bf16_wmma<<<dim3(4, 4, 8), blk, 0, stream>>>(
        xu, T, C * T,  wts, T, F * T,  512, 4096,
        nullptr, bm, C * F, (int)F,  /*bias_mode*/1, 0, 0,
        nullptr, out, F, 1, C * F);
}
